// HetGAT_37709812858999
// MI455X (gfx1250) — matrix operand order repaired and checked by reading, not compiled, not run on hardware
//
#include <hip/hip_runtime.h>
#include <hip/hip_bf16.h>
#include <cstddef>

#define NN 100000
#define NE 800000
#define ETOT (NN + NE)

typedef __attribute__((ext_vector_type(16))) _Float16 v16h;
typedef __attribute__((ext_vector_type(8)))  _Float16 v8h;
typedef __attribute__((ext_vector_type(8)))  float    v8f;

// ---------------------------------------------------------------------------
// WMMA GEMM: C[M,N] = act(A[M,K] @ W[K,N] + bias)
// fp32 in/out, f16 compute via v_wmma_f32_16x16x32_f16.
// Block = 256 threads = 8 wave32 waves. Tile: 64(M) x 128(N), K-step 32.
// Each wave owns a 16x... column strip and 4 M-subtiles -> 4 WMMAs / K-step.
// Interior tiles use a branch-free vectorized staging path (float4 loads,
// ds_store_b128); only edge blocks take the guarded scalar path.
// ---------------------------------------------------------------------------
__global__ void __launch_bounds__(256)
wmma_gemm_kernel(const float* __restrict__ A, const float* __restrict__ W,
                 const float* __restrict__ bias, float* __restrict__ C,
                 int M, int K, int N, int doRelu)
{
    __shared__ __align__(16) _Float16 ldsA[64 * 32];    // [m][k]
    __shared__ __align__(16) _Float16 ldsB[128 * 32];   // [n][k] (transposed)

    const int tid   = threadIdx.x;
    const int lane  = tid & 31;
    const int wave  = tid >> 5;
    const int l16   = lane & 15;
    const int hseg  = lane >> 4;          // 0 or 1 (lane half)
    const int mbase = blockIdx.y * 64;
    const int nbase = blockIdx.x * 128;

    // A staging coords: 8 contiguous k per thread
    const int arow = tid >> 2;            // 0..63
    const int acol = (tid & 3) * 8;       // 0,8,16,24

    v8f acc[4] = {};

    for (int k0 = 0; k0 < K; k0 += 32) {
        // ---- stage A tile 64x32 ----
        if (mbase + 64 <= M && k0 + 32 <= K) {
            const float* src = A + (size_t)(mbase + arow) * K + k0 + acol;
            float4 f0 = *(const float4*)(src);
            float4 f1 = *(const float4*)(src + 4);
            v8h h;
            h[0] = (_Float16)f0.x; h[1] = (_Float16)f0.y;
            h[2] = (_Float16)f0.z; h[3] = (_Float16)f0.w;
            h[4] = (_Float16)f1.x; h[5] = (_Float16)f1.y;
            h[6] = (_Float16)f1.z; h[7] = (_Float16)f1.w;
            *(v8h*)&ldsA[arow * 32 + acol] = h;
        } else {
            int gm = mbase + arow;
            v8h h;
#pragma unroll
            for (int j = 0; j < 8; ++j) {
                float f = 0.f;
                if (gm < M && (k0 + acol + j) < K)
                    f = A[(size_t)gm * K + k0 + acol + j];
                h[j] = (_Float16)f;
            }
            *(v8h*)&ldsA[arow * 32 + acol] = h;
        }
        // ---- stage W tile 32x128 transposed: ldsB[n][k] = W[k0+k][nbase+n] ----
        if (nbase + 128 <= N && k0 + 32 <= K) {
#pragma unroll
            for (int ch = 0; ch < 2; ++ch) {
                int chunk = tid + ch * 256;
                int n  = chunk & 127;
                int kc = (chunk >> 7) * 8;   // 0,8,16,24
                const float* wp = W + (size_t)(k0 + kc) * N + nbase + n;
                v8h h;
#pragma unroll
                for (int kk = 0; kk < 8; ++kk)
                    h[kk] = (_Float16)wp[(size_t)kk * N];
                *(v8h*)&ldsB[n * 32 + kc] = h;
            }
        } else {
#pragma unroll
            for (int ch = 0; ch < 2; ++ch) {
                int chunk = tid + ch * 256;
                int n  = chunk & 127;
                int kc = (chunk >> 7) * 8;
                v8h h;
#pragma unroll
                for (int kk = 0; kk < 8; ++kk) {
                    float f = 0.f;
                    if ((nbase + n) < N && (k0 + kc + kk) < K)
                        f = W[(size_t)(k0 + kc + kk) * N + nbase + n];
                    h[kk] = (_Float16)f;
                }
                *(v8h*)&ldsB[n * 32 + kc] = h;
            }
        }
        __syncthreads();

        // ---- B fragment (32x16): lane n = l16 (+wave strip); elem i<8 -> K = 8*hseg + i, i>=8 -> K = 16 + 8*hseg + (i-8)
        {
            int nrow = wave * 16 + l16;
            v8h b0 = *(const v8h*)&ldsB[nrow * 32 + 8 * hseg];
            v8h b1 = *(const v8h*)&ldsB[nrow * 32 + 16 + 8 * hseg];
            v16h bf;
#pragma unroll
            for (int i = 0; i < 8; ++i) { bf[i] = b0[i]; bf[i + 8] = b1[i]; }

            // ---- 4 A fragments / 4 WMMAs ----
#pragma unroll
            for (int mt = 0; mt < 4; ++mt) {
                int mrow = mt * 16 + l16;
                v8h a0 = *(const v8h*)&ldsA[mrow * 32 + 8 * hseg];
                v8h a1 = *(const v8h*)&ldsA[mrow * 32 + 16 + 8 * hseg];
                v16h af;
#pragma unroll
                for (int i = 0; i < 8; ++i) { af[i] = a0[i]; af[i + 8] = a1[i]; }
                acc[mt] = __builtin_amdgcn_wmma_f32_16x16x32_f16(
                              false, af, false, bf, (short)0, acc[mt], false, false);
            }
        }
        __syncthreads();
    }

    // ---- epilogue: C/D layout: lane n = l16; VGPR r -> m = r + 8*hseg ----
    int n = nbase + wave * 16 + l16;
    if (n < N) {
        float bv = bias ? bias[n] : 0.f;
#pragma unroll
        for (int mt = 0; mt < 4; ++mt) {
#pragma unroll
            for (int r = 0; r < 8; ++r) {
                int m = mbase + mt * 16 + r + 8 * hseg;
                if (m < M) {
                    float v = acc[mt][r] + bv;
                    if (doRelu) v = fmaxf(v, 0.f);
                    C[(size_t)m * N + n] = v;
                }
            }
        }
    }
}

// wa[i] = sum_j W[i*C + j] * v[j]   (small: one block)
__global__ void matvec_kernel(const float* __restrict__ W, const float* __restrict__ v,
                              float* __restrict__ out, int R, int Ccols)
{
    int r = blockIdx.x * blockDim.x + threadIdx.x;
    if (r >= R) return;
    float s = 0.f;
    for (int j = 0; j < Ccols; ++j) s += W[(size_t)r * Ccols + j] * v[j];
    out[r] = s;
}

// out[i] = dot(X[i,:], v) — one wave per row, shuffle reduction
__global__ void rowdot_kernel(const float* __restrict__ X, const float* __restrict__ v,
                              float* __restrict__ out, int Nrows, int C)
{
    int gw   = (int)((blockIdx.x * (size_t)blockDim.x + threadIdx.x) >> 5);
    int lane = threadIdx.x & 31;
    if (gw >= Nrows) return;
    const float* row = X + (size_t)gw * C;
    float s = 0.f;
    for (int j = lane; j < C; j += 32) s += row[j] * v[j];
#pragma unroll
    for (int off = 16; off > 0; off >>= 1) s += __shfl_down(s, off, 32);
    if (lane == 0) out[gw] = s;
}

__global__ void fill_kernel(float* __restrict__ p, float val, size_t n)
{
    size_t i = blockIdx.x * (size_t)blockDim.x + threadIdx.x;
    size_t stride = (size_t)gridDim.x * blockDim.x;
    for (; i < n; i += stride) p[i] = val;
}

__device__ __forceinline__ void atomicMaxF(float* addr, float val)
{
    if (val >= 0.f) atomicMax((int*)addr, __float_as_int(val));
    else            atomicMin((unsigned int*)addr, __float_as_uint(val));
}

__device__ __forceinline__ void edge_sd(const int* __restrict__ ei, int e, int& s, int& d)
{
    if (e < NE) { s = ei[e]; d = ei[NE + e]; }
    else        { s = d = e - NE; }
}

// pass 1: e = leaky_relu(a_s[src] + a_d[dst]); segment max over dst
__global__ void edge_max_kernel(const int* __restrict__ ei, const float* __restrict__ a_s,
                                const float* __restrict__ a_d, float* __restrict__ ebuf,
                                float* __restrict__ mx)
{
    int e = blockIdx.x * blockDim.x + threadIdx.x;
    if (e >= ETOT) return;
    int s, d; edge_sd(ei, e, s, d);
    float x = a_s[s] + a_d[d];
    x = (x > 0.f) ? x : 0.2f * x;
    ebuf[e] = x;
    atomicMaxF(&mx[d], x);
}

// pass 2: w = exp(e - m[dst]); segment sum over dst
__global__ void edge_exp_kernel(const int* __restrict__ ei, const float* __restrict__ mx,
                                float* __restrict__ ebuf, float* __restrict__ dn)
{
    int e = blockIdx.x * blockDim.x + threadIdx.x;
    if (e >= ETOT) return;
    int s, d; edge_sd(ei, e, s, d);
    float w = __expf(ebuf[e] - mx[d]);
    ebuf[e] = w;
    atomicAdd(&dn[d], w);
}

// pass 3: out[dst] += h_src[src] * alpha  — one block per edge, one thread/channel
__global__ void edge_scatter_kernel(const int* __restrict__ ei, const float* __restrict__ ebuf,
                                    const float* __restrict__ dn, const float* __restrict__ hsrc,
                                    float* __restrict__ out, int C)
{
    int e = blockIdx.x;
    int c = threadIdx.x;
    int s, d; edge_sd(ei, e, s, d);
    float alpha = ebuf[e] / (dn[d] + 1e-16f);
    atomicAdd(&out[(size_t)d * C + c], hsrc[(size_t)s * C + c] * alpha);
}

__global__ void bias_relu_kernel(float* __restrict__ X, const float* __restrict__ bias,
                                 size_t total, int C)
{
    size_t i = blockIdx.x * (size_t)blockDim.x + threadIdx.x;
    size_t stride = (size_t)gridDim.x * blockDim.x;
    for (; i < total; i += stride) {
        float v = X[i] + bias[i % C];
        X[i] = fmaxf(v, 0.f);
    }
}

// BN stats: stats[0..63] = sum, stats[64..127] = sum of squares
__global__ void bn_stats_kernel(const float* __restrict__ Y, float* __restrict__ stats,
                                int Nrows)
{
    __shared__ float ssum[64], ssq[64];
    int tid = threadIdx.x;
    if (tid < 64) { ssum[tid] = 0.f; ssq[tid] = 0.f; }
    __syncthreads();
    int c = tid & 63;
    int r0 = blockIdx.x * (blockDim.x >> 6) + (tid >> 6);
    int rstride = gridDim.x * (blockDim.x >> 6);
    float s = 0.f, q = 0.f;
    for (int r = r0; r < Nrows; r += rstride) {
        float v = Y[(size_t)r * 64 + c];
        s += v; q += v * v;
    }
    atomicAdd(&ssum[c], s);
    atomicAdd(&ssq[c], q);
    __syncthreads();
    if (tid < 64) {
        atomicAdd(&stats[tid], ssum[tid]);
        atomicAdd(&stats[64 + tid], ssq[tid]);
    }
}

// BN + relu + fc2(64x4) + log_softmax, one thread per node
__global__ void final_kernel(const float* __restrict__ Y, const float* __restrict__ stats,
                             const float* __restrict__ bn_g, const float* __restrict__ bn_b,
                             const float* __restrict__ fc2_w, const float* __restrict__ fc2_b,
                             float* __restrict__ out, int Nrows)
{
    __shared__ float sc[64], sh[64];
    int tid = threadIdx.x;
    if (tid < 64) {
        float invN = 1.f / (float)Nrows;
        float mu  = stats[tid] * invN;
        float var = stats[64 + tid] * invN - mu * mu;
        float is  = rsqrtf(var + 1e-5f) * bn_g[tid];
        sc[tid] = is;
        sh[tid] = bn_b[tid] - mu * is;
    }
    __syncthreads();
    int r = blockIdx.x * blockDim.x + tid;
    if (r >= Nrows) return;
    float acc0 = fc2_b[0], acc1 = fc2_b[1], acc2 = fc2_b[2], acc3 = fc2_b[3];
    const float* yr = Y + (size_t)r * 64;
#pragma unroll 4
    for (int c = 0; c < 64; ++c) {
        float v = fmaxf(yr[c] * sc[c] + sh[c], 0.f);
        acc0 += v * fc2_w[c * 4 + 0];
        acc1 += v * fc2_w[c * 4 + 1];
        acc2 += v * fc2_w[c * 4 + 2];
        acc3 += v * fc2_w[c * 4 + 3];
    }
    float m = fmaxf(fmaxf(acc0, acc1), fmaxf(acc2, acc3));
    float se = __expf(acc0 - m) + __expf(acc1 - m) + __expf(acc2 - m) + __expf(acc3 - m);
    float lse = m + __logf(se);
    out[(size_t)r * 4 + 0] = acc0 - lse;
    out[(size_t)r * 4 + 1] = acc1 - lse;
    out[(size_t)r * 4 + 2] = acc2 - lse;
    out[(size_t)r * 4 + 3] = acc3 - lse;
}

// ---------------------------------------------------------------------------
extern "C" void kernel_launch(void* const* d_in, const int* in_sizes, int n_in,
                              void* d_out, int out_size, void* d_ws, size_t ws_size,
                              hipStream_t stream)
{
    const float* x      = (const float*)d_in[0];
    const int*   ei     = (const int*)  d_in[1];
    const float* lin_w  = (const float*)d_in[2];
    const float* lin_b  = (const float*)d_in[3];
    const float* w1_src = (const float*)d_in[4];
    const float* w1_dst = (const float*)d_in[5];
    const float* a1_src = (const float*)d_in[6];
    const float* a1_dst = (const float*)d_in[7];
    const float* b1     = (const float*)d_in[8];
    const float* w2_src = (const float*)d_in[9];
    const float* w2_dst = (const float*)d_in[10];
    const float* a2_src = (const float*)d_in[11];
    const float* a2_dst = (const float*)d_in[12];
    const float* b2     = (const float*)d_in[13];
    const float* fc1_w  = (const float*)d_in[14];
    const float* fc1_b  = (const float*)d_in[15];
    const float* bn_g   = (const float*)d_in[16];
    const float* bn_b   = (const float*)d_in[17];
    const float* fc2_w  = (const float*)d_in[18];
    const float* fc2_b  = (const float*)d_in[19];
    float* out = (float*)d_out;

    // workspace layout (floats)
    float* ws    = (float*)d_ws;
    float* B0    = ws;                               // NN*256
    float* B1    = B0 + (size_t)NN * 256;            // NN*256
    float* a_s   = B1 + (size_t)NN * 256;            // NN
    float* a_d   = a_s + NN;                         // NN
    float* mx    = a_d + NN;                         // NN
    float* dn    = mx + NN;                          // NN
    float* ebuf  = dn + NN;                          // ETOT
    float* wa    = ebuf + (size_t)ETOT;              // 256
    float* stats = wa + 256;                         // 128

    const int egrid  = (ETOT + 255) / 256;
    const int rdgrid = (NN + 7) / 8;                 // 8 waves/block
    const int mgrid  = (NN + 63) / 64;               // GEMM row blocks

    // 1) h = relu(x @ lin_w + lin_b) -> B0   [NN x 256], K=300
    {
        dim3 g((256 + 127) / 128, mgrid);
        wmma_gemm_kernel<<<g, 256, 0, stream>>>(x, lin_w, lin_b, B0, NN, 300, 256, 1);
    }

    // ---------------- GAT layer 1 (C=256) ----------------
    {
        dim3 g((256 + 127) / 128, mgrid);
        wmma_gemm_kernel<<<g, 256, 0, stream>>>(B0, w1_src, nullptr, B1, NN, 256, 256, 0);
    }
    matvec_kernel<<<1, 256, 0, stream>>>(w1_dst, a1_dst, wa, 256, 256);
    rowdot_kernel<<<rdgrid, 256, 0, stream>>>(B1, a1_src, a_s, NN, 256);
    rowdot_kernel<<<rdgrid, 256, 0, stream>>>(B0, wa, a_d, NN, 256);
    fill_kernel<<<1024, 256, 0, stream>>>(mx, -1e30f, NN);
    fill_kernel<<<1024, 256, 0, stream>>>(dn, 0.f, NN);
    fill_kernel<<<4096, 256, 0, stream>>>(B0, 0.f, (size_t)NN * 256);
    edge_max_kernel<<<egrid, 256, 0, stream>>>(ei, a_s, a_d, ebuf, mx);
    edge_exp_kernel<<<egrid, 256, 0, stream>>>(ei, mx, ebuf, dn);
    edge_scatter_kernel<<<ETOT, 256, 0, stream>>>(ei, ebuf, dn, B1, B0, 256);
    bias_relu_kernel<<<4096, 256, 0, stream>>>(B0, b1, (size_t)NN * 256, 256);

    // ---------------- GAT layer 2 (C=128) ----------------
    {
        dim3 g((128 + 127) / 128, mgrid);
        wmma_gemm_kernel<<<g, 256, 0, stream>>>(B0, w2_src, nullptr, B1, NN, 256, 128, 0);
    }
    matvec_kernel<<<1, 256, 0, stream>>>(w2_dst, a2_dst, wa, 256, 128);
    rowdot_kernel<<<rdgrid, 256, 0, stream>>>(B1, a2_src, a_s, NN, 128);
    rowdot_kernel<<<rdgrid, 256, 0, stream>>>(B0, wa, a_d, NN, 256);
    fill_kernel<<<1024, 256, 0, stream>>>(mx, -1e30f, NN);
    fill_kernel<<<1024, 256, 0, stream>>>(dn, 0.f, NN);
    fill_kernel<<<4096, 256, 0, stream>>>(B0, 0.f, (size_t)NN * 128);
    edge_max_kernel<<<egrid, 256, 0, stream>>>(ei, a_s, a_d, ebuf, mx);
    edge_exp_kernel<<<egrid, 256, 0, stream>>>(ei, mx, ebuf, dn);
    edge_scatter_kernel<<<ETOT, 128, 0, stream>>>(ei, ebuf, dn, B1, B0, 128);
    bias_relu_kernel<<<4096, 256, 0, stream>>>(B0, b2, (size_t)NN * 128, 128);

    // ---------------- head ----------------
    // fc1: B0[NN,128] @ fc1_w[128,64] + fc1_b -> B1[NN,64]
    {
        dim3 g((64 + 127) / 128, mgrid);
        wmma_gemm_kernel<<<g, 256, 0, stream>>>(B0, fc1_w, fc1_b, B1, NN, 128, 64, 0);
    }
    fill_kernel<<<1, 256, 0, stream>>>(stats, 0.f, 128);
    bn_stats_kernel<<<1024, 256, 0, stream>>>(B1, stats, NN);
    final_kernel<<<(NN + 255) / 256, 256, 0, stream>>>(B1, stats, bn_g, bn_b,
                                                       fc2_w, fc2_b, out, NN);
}
